// LinearAttentionChunkLayer_7834020348443
// MI455X (gfx1250) — matrix operand order repaired and checked by reading, not compiled, not run on hardware
//
#include <hip/hip_runtime.h>

#define B_   4
#define T_   2048
#define D_   1024
#define C_   128
#define NCH_ (T_ / C_)
#define BT_  (B_ * T_)

typedef __attribute__((ext_vector_type(16))) _Float16 v16h;
typedef __attribute__((ext_vector_type(8)))  _Float16 v8h;
typedef __attribute__((ext_vector_type(8)))  float    v8f;

// ---------------------------------------------------------------------------
// CDNA5 async global->LDS copy (ASYNCcnt-tracked, CDNA5 ISA)
// Probe-derived builtin signature: (int4 AS(1)* src, int4 AS(3)* dst, Ii, Ii)
// ---------------------------------------------------------------------------
#if __has_builtin(__builtin_amdgcn_global_load_async_to_lds_b128) && \
    __has_builtin(__builtin_amdgcn_s_wait_asynccnt)
#define HAS_ASYNC_LDS 1
#endif

typedef int v4i_ __attribute__((vector_size(16)));
typedef __attribute__((address_space(1))) v4i_ g_v4i;
typedef __attribute__((address_space(3))) v4i_ l_v4i;

// copy 16 bytes (8 f16) global -> LDS, asynchronously when supported
__device__ __forceinline__ void async_cp16(const _Float16* g, _Float16* l) {
#ifdef HAS_ASYNC_LDS
    __builtin_amdgcn_global_load_async_to_lds_b128((g_v4i*)g, (l_v4i*)l, 0, 0);
#else
    *(v8h*)l = *(const v8h*)g;
#endif
}

__device__ __forceinline__ void wait_async0() {
#ifdef HAS_ASYNC_LDS
    __builtin_amdgcn_s_wait_asynccnt(0);
#endif
}

// ---------------------------------------------------------------------------
// Fragment helpers (CDNA5 wave32 WMMA 16x16x32 f16 layouts, CDNA5 ISA)
// ---------------------------------------------------------------------------

// A: 16x32 f16, row-major source. Lane l holds row (l&15); K elements
// [h*8 .. h*8+7] and [16+h*8 .. 16+h*8+7] with h = l>>4.
__device__ __forceinline__ v16h load_a_frag(const _Float16* tile, int ld, int lane) {
    const int m = lane & 15;
    const int h = lane >> 4;
    const _Float16* p = tile + (size_t)m * ld + h * 8;
    v8h lo = *(const v8h*)(p);
    v8h hi = *(const v8h*)(p + 16);
    return __builtin_shufflevector(lo, hi, 0, 1, 2, 3, 4, 5, 6, 7,
                                           8, 9, 10, 11, 12, 13, 14, 15);
}

// B: 32x16 f16 where the operand is stored transposed [N][K] (K contiguous).
// Lane l holds column n=(l&15), K elements [h*8 .. h*8+7] and [16+h*8 .. 16+h*8+7], h = l>>4.
__device__ __forceinline__ v16h load_b_frag(const _Float16* tileT, int ld, int lane) {
    const int n = lane & 15;
    const int h = lane >> 4;
    const _Float16* p = tileT + (size_t)n * ld + h * 8;
    v8h lo = *(const v8h*)(p);
    v8h hi = *(const v8h*)(p + 16);
    return __builtin_shufflevector(lo, hi, 0, 1, 2, 3, 4, 5, 6, 7,
                                           8, 9, 10, 11, 12, 13, 14, 15);
}

__device__ __forceinline__ v8f wmma16(v16h a, v16h b, v8f c) {
    return __builtin_amdgcn_wmma_f32_16x16x32_f16(false, a, false, b,
                                                  (short)0, c, false, false);
}

// ---------------------------------------------------------------------------
// Utility kernels
// ---------------------------------------------------------------------------

__global__ void cvt_f32_f16(const float* __restrict__ in, _Float16* __restrict__ out, size_t n) {
    size_t i = (size_t)blockIdx.x * blockDim.x + threadIdx.x;
    size_t stride = (size_t)gridDim.x * blockDim.x;
    for (; i < n; i += stride) out[i] = (_Float16)in[i];
}

__global__ void zero_u32(unsigned int* __restrict__ p, size_t n) {
    size_t i = (size_t)blockIdx.x * blockDim.x + threadIdx.x;
    size_t stride = (size_t)gridDim.x * blockDim.x;
    for (; i < n; i += stride) p[i] = 0u;
}

// [B,T,D] f16 -> [B,D,T] f16 via LDS tiles.
__global__ void transpose_bt(const _Float16* __restrict__ in, _Float16* __restrict__ out) {
    __shared__ _Float16 tile[32][33];
    const int b = blockIdx.z;
    const int col = blockIdx.x * 32 + threadIdx.x;  // d
#pragma unroll
    for (int r = 0; r < 4; ++r) {
        int row = blockIdx.y * 32 + threadIdx.y + r * 8;  // t
        tile[threadIdx.y + r * 8][threadIdx.x] = in[((size_t)b * T_ + row) * D_ + col];
    }
    __syncthreads();
#pragma unroll
    for (int r = 0; r < 4; ++r) {
        int d = blockIdx.x * 32 + threadIdx.y + r * 8;
        int t = blockIdx.y * 32 + threadIdx.x;
        out[((size_t)b * D_ + d) * T_ + t] = tile[threadIdx.x][threadIdx.y + r * 8];
    }
}

// ---------------------------------------------------------------------------
// Generic GEMM: out = act(A[M,K] @ Bt[N,K]^T + bias), double-buffered async
// global->LDS staging. Block = 128x128 tile, 4 waves, wave = 64x64 (4x4 WMMA).
// __launch_bounds__(128, 1): one wave per SIMD -> full VGPR file, no spills.
// ---------------------------------------------------------------------------
__global__ void __launch_bounds__(128, 1)
gemm_tn(const _Float16* __restrict__ A, const _Float16* __restrict__ Bt,
        const float* __restrict__ bias,
        _Float16* __restrict__ outH, float* __restrict__ outF,
        int M, int N, int K, int fm) {
    (void)M;
    __shared__ _Float16 Ab[2][128][32];
    __shared__ _Float16 Bb[2][128][32];

    const int t = threadIdx.x;            // 0..127
    const int lane = t & 31;
    const int w = t >> 5;
    const int wr = w >> 1, wc = w & 1;
    const int row0 = blockIdx.y * 128 + wr * 64;
    const int col0 = blockIdx.x * 128 + wc * 64;

    // one global row per thread for each operand
    const _Float16* gA = A + (size_t)(blockIdx.y * 128 + t) * K;
    const _Float16* gB = Bt + (size_t)(blockIdx.x * 128 + t) * K;

    // stage a 128x32 k-slab of A and B into LDS buffer bf (4 x b128 per row)
    auto stage = [&](int bf, int k0) {
#pragma unroll
        for (int q = 0; q < 4; ++q) {
            async_cp16(gA + k0 + q * 8, &Ab[bf][t][q * 8]);
            async_cp16(gB + k0 + q * 8, &Bb[bf][t][q * 8]);
        }
    };

    stage(0, 0);
    wait_async0();
    __syncthreads();

    v8f c[4][4] = {};
    int buf = 0;
#pragma unroll 1
    for (int k = 0; k < K; k += 32) {
        if (k + 32 < K) stage(buf ^ 1, k + 32);  // DMA overlaps the WMMA block

        v16h a[4], bfr[4];
#pragma unroll
        for (int i = 0; i < 4; ++i)
            a[i] = load_a_frag(&Ab[buf][wr * 64 + i * 16][0], 32, lane);
#pragma unroll
        for (int j = 0; j < 4; ++j)
            bfr[j] = load_b_frag(&Bb[buf][wc * 64 + j * 16][0], 32, lane);
#pragma unroll
        for (int i = 0; i < 4; ++i)
#pragma unroll
            for (int j = 0; j < 4; ++j)
                c[i][j] = wmma16(a[i], bfr[j], c[i][j]);

        wait_async0();
        __syncthreads();
        buf ^= 1;
    }

    const int n = lane & 15, hh = lane >> 4;
#pragma unroll
    for (int i = 0; i < 4; ++i)
#pragma unroll
        for (int j = 0; j < 4; ++j)
#pragma unroll
            for (int r = 0; r < 8; ++r) {
                int row = row0 + i * 16 + hh * 8 + r;
                int col = col0 + j * 16 + n;
                float v = c[i][j][r] + bias[col];
                if (fm) v = (v > 0.0f) ? (v + 1.0f) : __expf(v);
                if (outH) outH[(size_t)row * N + col] = (_Float16)v;
                if (outF) outF[(size_t)row * N + col] = v;
            }
}

// ---------------------------------------------------------------------------
// Per-chunk kernel 1: S = tril(q_i k_i^T) (f16 to ws), rowsum, and
// den = rowsum + q_i . key_sum + eps. One block (128 thr) per batch.
// ---------------------------------------------------------------------------
__global__ void __launch_bounds__(128, 1)
attn_scores(const _Float16* __restrict__ Qb, const _Float16* __restrict__ Kb,
            const float* __restrict__ key_sum,
            _Float16* __restrict__ Sbuf, float* __restrict__ den, int ch) {
    const int b = blockIdx.x;
    const int lane = threadIdx.x & 31;
    const int w = threadIdx.x >> 5;
    const int wr = w >> 1, wc = w & 1;
    const int row0 = wr * 64, col0 = wc * 64;

    __shared__ float rs[C_];
    rs[threadIdx.x] = 0.0f;
    __syncthreads();

    const _Float16* Abase = Qb + ((size_t)b * T_ + (size_t)ch * C_) * D_;
    const _Float16* Bbase = Kb + ((size_t)b * T_ + (size_t)ch * C_) * D_;

    v8f c[4][4] = {};
#pragma unroll 1
    for (int k = 0; k < D_; k += 32) {
        v16h a[4], bfr[4];
#pragma unroll
        for (int i = 0; i < 4; ++i)
            a[i] = load_a_frag(Abase + (size_t)(row0 + i * 16) * D_ + k, D_, lane);
#pragma unroll
        for (int j = 0; j < 4; ++j)
            bfr[j] = load_b_frag(Bbase + (size_t)(col0 + j * 16) * D_ + k, D_, lane);
#pragma unroll
        for (int i = 0; i < 4; ++i)
#pragma unroll
            for (int j = 0; j < 4; ++j)
                c[i][j] = wmma16(a[i], bfr[j], c[i][j]);
    }

    const int n = lane & 15, hh = lane >> 4;
#pragma unroll
    for (int i = 0; i < 4; ++i)
#pragma unroll
        for (int j = 0; j < 4; ++j)
#pragma unroll
            for (int r = 0; r < 8; ++r) {
                int rr = row0 + i * 16 + hh * 8 + r;  // query index c
                int ss = col0 + j * 16 + n;           // key index s
                float v = (ss <= rr) ? c[i][j][r] : 0.0f;  // causal (inclusive) mask
                atomicAdd(&rs[rr], v);
                Sbuf[((size_t)b * C_ + rr) * C_ + ss] = (_Float16)v;
            }
    __syncthreads();

    // den[c] = rowsum + q[c] . key_sum + eps  (one thread per row)
    const int t = threadIdx.x;
    const _Float16* qrow = Qb + ((size_t)b * T_ + (size_t)ch * C_ + t) * D_;
    const float* ks = key_sum + (size_t)b * D_;
    float dot = 0.0f;
    for (int d = 0; d < D_; ++d) dot += (float)qrow[d] * ks[d];
    den[b * C_ + t] = rs[t] + dot + 1e-6f;
}

// ---------------------------------------------------------------------------
// Per-chunk kernel 2: num = S @ v_i + q_i @ KV ; attn = num / den (f16 to ws).
// Grid (D/128 col slabs, B). KVB is the f16 mirror of KV^T ([e][d], d contig).
// ---------------------------------------------------------------------------
__global__ void __launch_bounds__(128, 1)
attn_out(const _Float16* __restrict__ Qb, const _Float16* __restrict__ Sbuf,
         const _Float16* __restrict__ Vt, const _Float16* __restrict__ KVB,
         const float* __restrict__ den, _Float16* __restrict__ Attn, int ch) {
    const int b = blockIdx.y;
    const int lane = threadIdx.x & 31;
    const int w = threadIdx.x >> 5;
    const int wr = w >> 1, wc = w & 1;
    const int row0 = wr * 64;                              // chunk-local row c
    const int col0 = blockIdx.x * 128 + wc * 64;           // value dim e

    v8f c[4][4] = {};

    // (1) intra-chunk: S[c][s] @ v[s][e]  (B operand = Vt rows e, contiguous s)
    const _Float16* Sb = Sbuf + (size_t)b * C_ * C_;
    const _Float16* Vb = Vt + (size_t)b * D_ * T_ + (size_t)ch * C_;
#pragma unroll 1
    for (int k = 0; k < C_; k += 32) {
        v16h a[4], bfr[4];
#pragma unroll
        for (int i = 0; i < 4; ++i)
            a[i] = load_a_frag(Sb + (size_t)(row0 + i * 16) * C_ + k, C_, lane);
#pragma unroll
        for (int j = 0; j < 4; ++j)
            bfr[j] = load_b_frag(Vb + (size_t)(col0 + j * 16) * T_ + k, T_, lane);
#pragma unroll
        for (int i = 0; i < 4; ++i)
#pragma unroll
            for (int j = 0; j < 4; ++j)
                c[i][j] = wmma16(a[i], bfr[j], c[i][j]);
    }

    // (2) inter-chunk: q[c][d] @ KV[d][e]  (B operand = KVB rows e, contiguous d)
    const _Float16* Qbase = Qb + ((size_t)b * T_ + (size_t)ch * C_) * D_;
    const _Float16* KVb = KVB + (size_t)b * D_ * D_;
#pragma unroll 1
    for (int k = 0; k < D_; k += 32) {
        v16h a[4], bfr[4];
#pragma unroll
        for (int i = 0; i < 4; ++i)
            a[i] = load_a_frag(Qbase + (size_t)(row0 + i * 16) * D_ + k, D_, lane);
#pragma unroll
        for (int j = 0; j < 4; ++j)
            bfr[j] = load_b_frag(KVb + (size_t)(col0 + j * 16) * D_ + k, D_, lane);
#pragma unroll
        for (int i = 0; i < 4; ++i)
#pragma unroll
            for (int j = 0; j < 4; ++j)
                c[i][j] = wmma16(a[i], bfr[j], c[i][j]);
    }

    const int n = lane & 15, hh = lane >> 4;
#pragma unroll
    for (int i = 0; i < 4; ++i)
#pragma unroll
        for (int j = 0; j < 4; ++j)
#pragma unroll
            for (int r = 0; r < 8; ++r) {
                int rr = row0 + i * 16 + hh * 8 + r;
                int e = col0 + j * 16 + n;
                float v = c[i][j][r] / den[b * C_ + rr];
                Attn[((size_t)b * T_ + (size_t)ch * C_ + rr) * D_ + e] = (_Float16)v;
            }
}

// ---------------------------------------------------------------------------
// Per-chunk kernel 3: kv[d][e] += k_i^T v_i (f32 master, lives in d_out),
// KVB[e][d] = f16(kv) mirror, key_sum[d] += sum_c k_i[c][d].
// Grid (D/128 d-tiles, D/128 e-tiles, B).
// ---------------------------------------------------------------------------
__global__ void __launch_bounds__(128, 1)
state_update(const _Float16* __restrict__ Kt, const _Float16* __restrict__ Vt,
             float* __restrict__ kv, float* __restrict__ key_sum,
             _Float16* __restrict__ KVB, int ch) {
    const int b = blockIdx.z;
    const int lane = threadIdx.x & 31;
    const int w = threadIdx.x >> 5;
    const int wr = w >> 1, wc = w & 1;
    const int d0 = blockIdx.x * 128 + wr * 64;
    const int e0 = blockIdx.y * 128 + wc * 64;

    const _Float16* Kb = Kt + (size_t)b * D_ * T_ + (size_t)ch * C_;  // row d, contig c
    const _Float16* Vb = Vt + (size_t)b * D_ * T_ + (size_t)ch * C_;  // row e, contig c

    v8f c[4][4] = {};
#pragma unroll 1
    for (int k = 0; k < C_; k += 32) {
        v16h a[4], bfr[4];
#pragma unroll
        for (int i = 0; i < 4; ++i)
            a[i] = load_a_frag(Kb + (size_t)(d0 + i * 16) * T_ + k, T_, lane);
#pragma unroll
        for (int j = 0; j < 4; ++j)
            bfr[j] = load_b_frag(Vb + (size_t)(e0 + j * 16) * T_ + k, T_, lane);
#pragma unroll
        for (int i = 0; i < 4; ++i)
#pragma unroll
            for (int j = 0; j < 4; ++j)
                c[i][j] = wmma16(a[i], bfr[j], c[i][j]);
    }

    const int n = lane & 15, hh = lane >> 4;
#pragma unroll
    for (int i = 0; i < 4; ++i)
#pragma unroll
        for (int j = 0; j < 4; ++j)
#pragma unroll
            for (int r = 0; r < 8; ++r) {
                int d = d0 + i * 16 + hh * 8 + r;
                int e = e0 + j * 16 + n;
                float* kp = kv + ((size_t)b * D_ + d) * D_ + e;
                float nv = *kp + c[i][j][r];
                *kp = nv;
                KVB[((size_t)b * D_ + e) * D_ + d] = (_Float16)nv;
            }

    if (blockIdx.y == 0) {
        const int d = blockIdx.x * 128 + threadIdx.x;
        const _Float16* kr = Kt + (size_t)b * D_ * T_ + (size_t)d * T_ + (size_t)ch * C_;
        float s = 0.0f;
        for (int cc = 0; cc < C_; ++cc) s += (float)kr[cc];
        key_sum[(size_t)b * D_ + d] += s;
    }
}

// ---------------------------------------------------------------------------
// Host orchestration
// ---------------------------------------------------------------------------
extern "C" void kernel_launch(void* const* d_in, const int* in_sizes, int n_in,
                              void* d_out, int out_size, void* d_ws, size_t ws_size,
                              hipStream_t stream) {
    (void)in_sizes; (void)n_in; (void)out_size; (void)ws_size;

    const float* chunk = (const float*)d_in[0];
    const float* Wq = (const float*)d_in[1]; const float* bq = (const float*)d_in[2];
    const float* Wk = (const float*)d_in[3]; const float* bk = (const float*)d_in[4];
    const float* Wv = (const float*)d_in[5]; const float* bv = (const float*)d_in[6];
    const float* Wo = (const float*)d_in[7]; const float* bo = (const float*)d_in[8];

    float* outF = (float*)d_out;                              // [B,T,D]
    float* key_sum = outF + (size_t)BT_ * D_;                 // [B,D]
    float* kv = key_sum + (size_t)B_ * D_;                    // [B,D,D]

    char* ws = (char*)d_ws;
    size_t off = 0;
    auto take = [&](size_t bytes) -> char* {
        char* p = ws + off;
        off += (bytes + 255) & ~(size_t)255;
        return p;
    };
    _Float16* Xb   = (_Float16*)take((size_t)BT_ * D_ * 2);
    _Float16* Wqb  = (_Float16*)take((size_t)D_ * D_ * 2);
    _Float16* Wkb  = (_Float16*)take((size_t)D_ * D_ * 2);
    _Float16* Wvb  = (_Float16*)take((size_t)D_ * D_ * 2);
    _Float16* Wob  = (_Float16*)take((size_t)D_ * D_ * 2);
    _Float16* Qb   = (_Float16*)take((size_t)BT_ * D_ * 2);
    _Float16* Kb   = (_Float16*)take((size_t)BT_ * D_ * 2);
    _Float16* Vb   = (_Float16*)take((size_t)BT_ * D_ * 2);
    _Float16* Kt   = (_Float16*)take((size_t)BT_ * D_ * 2);
    _Float16* Vt   = (_Float16*)take((size_t)BT_ * D_ * 2);
    _Float16* Attn = (_Float16*)take((size_t)BT_ * D_ * 2);
    _Float16* KVB  = (_Float16*)take((size_t)B_ * D_ * D_ * 2);
    _Float16* Sbuf = (_Float16*)take((size_t)B_ * C_ * C_ * 2);
    float*    den  = (float*)take((size_t)B_ * C_ * 4);

    const dim3 blk(128);

    // f32 -> f16 staging
    cvt_f32_f16<<<512, 256, 0, stream>>>(chunk, Xb, (size_t)BT_ * D_);
    cvt_f32_f16<<<256, 256, 0, stream>>>(Wq, Wqb, (size_t)D_ * D_);
    cvt_f32_f16<<<256, 256, 0, stream>>>(Wk, Wkb, (size_t)D_ * D_);
    cvt_f32_f16<<<256, 256, 0, stream>>>(Wv, Wvb, (size_t)D_ * D_);
    cvt_f32_f16<<<256, 256, 0, stream>>>(Wo, Wob, (size_t)D_ * D_);

    // zero recurrent state: key_sum + kv (contiguous in d_out) and the f16 mirror
    zero_u32<<<512, 256, 0, stream>>>((unsigned int*)key_sum,
                                      (size_t)B_ * D_ + (size_t)B_ * D_ * D_);
    zero_u32<<<512, 256, 0, stream>>>((unsigned int*)KVB, (size_t)B_ * D_ * D_ / 2);

    // projections (elu+1 fused for q,k)
    const dim3 ggrid(D_ / 128, BT_ / 128);
    gemm_tn<<<ggrid, blk, 0, stream>>>(Xb, Wqb, bq, Qb, nullptr, BT_, D_, D_, 1);
    gemm_tn<<<ggrid, blk, 0, stream>>>(Xb, Wkb, bk, Kb, nullptr, BT_, D_, D_, 1);
    gemm_tn<<<ggrid, blk, 0, stream>>>(Xb, Wvb, bv, Vb, nullptr, BT_, D_, D_, 0);

    // transposed mirrors for WMMA-B operands
    const dim3 tgrid(D_ / 32, T_ / 32, B_), tblk(32, 8);
    transpose_bt<<<tgrid, tblk, 0, stream>>>(Kb, Kt);
    transpose_bt<<<tgrid, tblk, 0, stream>>>(Vb, Vt);

    // chunked causal linear-attention recurrence (serialized on stream)
    for (int ch = 0; ch < NCH_; ++ch) {
        attn_scores<<<dim3(B_), blk, 0, stream>>>(Qb, Kb, key_sum, Sbuf, den, ch);
        attn_out<<<dim3(D_ / 128, B_), blk, 0, stream>>>(Qb, Sbuf, Vt, KVB, den, Attn, ch);
        state_update<<<dim3(D_ / 128, D_ / 128, B_), blk, 0, stream>>>(Kt, Vt, kv, key_sum,
                                                                       KVB, ch);
    }

    // output projection with bias, f32 result straight into d_out
    gemm_tn<<<ggrid, blk, 0, stream>>>(Attn, Wob, bo, nullptr, outF, BT_, D_, D_, 0);
}
